// FDSECNN_29764123361285
// MI455X (gfx1250) — matrix operand order repaired and checked by reading, not compiled, not run on hardware
//
#include <hip/hip_runtime.h>
#include <hip/hip_bf16.h>

// ---------------------------------------------------------------------------
// MI455X (gfx1250, wave32). Memory-bound CNN pipeline; WMMA f16 for all
// GEMM-shaped work; async-to-LDS for conv2 patch staging.
// ---------------------------------------------------------------------------

typedef __attribute__((ext_vector_type(16))) _Float16 v16h;
typedef __attribute__((ext_vector_type(8)))  _Float16 v8h;
typedef __attribute__((ext_vector_type(8)))  float    v8f;

#define EPSV 1e-5f

// ------------------------------- utility kernels ---------------------------

__global__ void kZero(float* p, int n) {
    int i = blockIdx.x * blockDim.x + threadIdx.x;
    if (i < n) p[i] = 0.0f;
}

__global__ void kCvtF16(const float* __restrict__ in, _Float16* __restrict__ out, int n) {
    int i = blockIdx.x * blockDim.x + threadIdx.x;
    if (i < n) out[i] = (_Float16)in[i];
}

// fc3_w [10,512] -> padded f16 [32,512]
__global__ void kCvtFc3(const float* __restrict__ in, _Float16* __restrict__ out) {
    int i = blockIdx.x * blockDim.x + threadIdx.x;
    if (i < 32 * 512) {
        int nrow = i >> 9, k = i & 511;
        out[i] = (nrow < 10) ? (_Float16)in[nrow * 512 + k] : (_Float16)0.0f;
    }
}

// dfe2_w [16,16,3,3] -> w2t f16 [16 out-ch][K=160] (K padded 144->160 with 0)
__global__ void kCvtW2(const float* __restrict__ in, _Float16* __restrict__ out) {
    int i = blockIdx.x * blockDim.x + threadIdx.x;
    if (i < 16 * 160) {
        int o = i / 160, k = i % 160;
        out[i] = (k < 144) ? (_Float16)in[o * 144 + k] : (_Float16)0.0f;
    }
}

// ------------------------------- conv1 (direct) ----------------------------
// x [B,3,32,32], w [8,3,3,3] -> y [B,8,32,32]. One thread computes all 8 oc
// for one (b,oy,ox): 27 input loads amortized 8x; weights staged in LDS.
__global__ void kConv1(const float* __restrict__ x, const float* __restrict__ w,
                       float* __restrict__ y) {
    __shared__ float sw[216];
    if (threadIdx.x < 216) sw[threadIdx.x] = w[threadIdx.x];
    __syncthreads();
    size_t idx = (size_t)blockIdx.x * blockDim.x + threadIdx.x;  // B*1024 threads
    int ox = idx & 31, oy = (idx >> 5) & 31;
    size_t b = idx >> 10;
    float in[3][3][3];
    #pragma unroll
    for (int ic = 0; ic < 3; ++ic) {
        const float* xp = x + ((b * 3 + ic) << 10);
        #pragma unroll
        for (int ky = 0; ky < 3; ++ky) {
            int iy = oy - 1 + ky;
            #pragma unroll
            for (int kx = 0; kx < 3; ++kx) {
                int ix = ox - 1 + kx;
                bool ok = (iy >= 0) && (iy < 32) && (ix >= 0) && (ix < 32);
                in[ic][ky][kx] = ok ? xp[iy * 32 + ix] : 0.0f;
            }
        }
    }
    #pragma unroll
    for (int oc = 0; oc < 8; ++oc) {
        float acc = 0.0f;
        #pragma unroll
        for (int ic = 0; ic < 3; ++ic)
            #pragma unroll
            for (int ky = 0; ky < 3; ++ky)
                #pragma unroll
                for (int kx = 0; kx < 3; ++kx)
                    acc += in[ic][ky][kx] * sw[((oc * 3 + ic) * 3 + ky) * 3 + kx];
        y[((b * 8 + oc) << 10) + (oy << 5) + ox] = acc;
    }
}

// ------------------------------- BN batch-stat reduce ----------------------
// x [Bn,C,S], S = 1<<sshift. blockIdx.y = channel. float4 loads, shift math.
__global__ void kStats(const float* __restrict__ x, int C, int sshift, size_t nvec,
                       float* __restrict__ sumOut, float* __restrict__ sqOut) {
    int c = blockIdx.y;
    int vshift = sshift - 2;
    size_t vmask = ((size_t)1 << vshift) - 1;
    size_t stride = (size_t)gridDim.x * blockDim.x;
    float s = 0.0f, q = 0.0f;
    for (size_t i = (size_t)blockIdx.x * blockDim.x + threadIdx.x; i < nvec; i += stride) {
        size_t b = i >> vshift;
        size_t v = i & vmask;
        const float4* p = (const float4*)(x + ((b * C + c) << sshift)) + v;
        float4 t = *p;
        s += t.x + t.y + t.z + t.w;
        q += t.x * t.x + t.y * t.y + t.z * t.z + t.w * t.w;
    }
    __shared__ float rs[256], rq[256];
    rs[threadIdx.x] = s; rq[threadIdx.x] = q;
    __syncthreads();
    for (int st = 128; st > 0; st >>= 1) {
        if ((int)threadIdx.x < st) {
            rs[threadIdx.x] += rs[threadIdx.x + st];
            rq[threadIdx.x] += rq[threadIdx.x + st];
        }
        __syncthreads();
    }
    if (threadIdx.x == 0) {
        atomicAdd(&sumOut[c], rs[0]);
        atomicAdd(&sqOut[c], rq[0]);
    }
}

__global__ void kFinalize(const float* __restrict__ sum, const float* __restrict__ sq,
                          float* __restrict__ mean, float* __restrict__ var,
                          int C, float invN) {
    int c = threadIdx.x;
    if (c < C) {
        float m = sum[c] * invN;
        mean[c] = m;
        var[c] = sq[c] * invN - m * m;
    }
}

// normalize + relu, in place, float4.
__global__ void kNorm(float* __restrict__ x, int C, int sshift, size_t nvec,
                      const float* __restrict__ mean, const float* __restrict__ var,
                      const float* __restrict__ g, const float* __restrict__ bta) {
    int c = blockIdx.y;
    float m = mean[c];
    float sc = rsqrtf(var[c] + EPSV) * g[c];
    float sh = bta[c] - m * sc;
    int vshift = sshift - 2;
    size_t vmask = ((size_t)1 << vshift) - 1;
    size_t stride = (size_t)gridDim.x * blockDim.x;
    for (size_t i = (size_t)blockIdx.x * blockDim.x + threadIdx.x; i < nvec; i += stride) {
        size_t b = i >> vshift;
        size_t v = i & vmask;
        float4* p = (float4*)(x + ((b * C + c) << sshift)) + v;
        float4 t = *p;
        t.x = fmaxf(t.x * sc + sh, 0.0f);
        t.y = fmaxf(t.y * sc + sh, 0.0f);
        t.z = fmaxf(t.z * sc + sh, 0.0f);
        t.w = fmaxf(t.w * sc + sh, 0.0f);
        *p = t;
    }
}

// ------------------------- expand + BN2 + relu + maxpool -------------------
// r [Bn,C,H,H], H = 2<<pshift. Expanded channel (i,g) = w*r -> batch stats
// are mean=w*m_i, var=w^2*v_i exactly. out f16 [Bn, C*G, H/2, H/2].
__global__ void kExpandPool(const float* __restrict__ r, const float* __restrict__ wse,
                            const float* __restrict__ rmean, const float* __restrict__ rvar,
                            const float* __restrict__ bng, const float* __restrict__ bnb,
                            _Float16* __restrict__ out, int C, int pshift, int chshift,
                            size_t total) {
    int P = 1 << pshift;           // pooled width
    int H = P << 1;
    size_t stride = (size_t)gridDim.x * blockDim.x;
    for (size_t idx = (size_t)blockIdx.x * blockDim.x + threadIdx.x; idx < total; idx += stride) {
        int px = (int)(idx & (P - 1));
        int py = (int)((idx >> pshift) & (P - 1));
        int ch = (int)((idx >> (2 * pshift)) & ((1 << chshift) - 1));
        size_t b = idx >> (2 * pshift + chshift);
        int i = ch >> 1;           // G == 2
        float w = wse[ch];
        float inv = rsqrtf(w * w * rvar[i] + EPSV);
        float gsc = bng[ch];
        float sc = w * inv * gsc;
        float sh = bnb[ch] - w * rmean[i] * inv * gsc;
        const float2* rp = (const float2*)(r + ((b * C + i) << (2 * pshift + 2)));
        int rowv = H >> 1;         // float2 per row
        float2 t0 = rp[(2 * py) * rowv + px];
        float2 t1 = rp[(2 * py + 1) * rowv + px];
        float e0 = fmaxf(sc * t0.x + sh, 0.0f);
        float e1 = fmaxf(sc * t0.y + sh, 0.0f);
        float e2 = fmaxf(sc * t1.x + sh, 0.0f);
        float e3 = fmaxf(sc * t1.y + sh, 0.0f);
        out[idx] = (_Float16)fmaxf(fmaxf(e0, e1), fmaxf(e2, e3));
    }
}

// ------------------------------- conv2: implicit GEMM WMMA -----------------
// a1 f16 [B,16,16,16]; w2t f16 [16][160]; y2 f32 [B,16,16,16].
// One wave per (b,oy). Stage 1: async-copy 16ch x 3row patch to LDS
// (rows stride 20 halfs, interior at half-index 2..17 so every B32 async
// transfer is 4B aligned on both ends; pads pre-zeroed via ds_store).
// Stage 2: expand patch -> per-wave im2col tile Aim[16 ox][160 K] in LDS with
// fully-unrolled constant indexing (no div/mod). Stage 3: MMA loop is pure
// ds_load_b128 A-frags + global b128 B-frags + v_wmma.
__global__ void kConv2Wmma(const _Float16* __restrict__ a1,
                           const _Float16* __restrict__ w2t,
                           float* __restrict__ y2) {
    __shared__ _Float16 patch[4][960];    // 48 rows * 20 halfs per wave
    __shared__ _Float16 Aim[4][2560];     // 16 ox * 160 K per wave
    int wave = threadIdx.x >> 5, lane = threadIdx.x & 31;
    int tile = blockIdx.x * 4 + wave;     // = b*16 + oy
    int bb = tile >> 4, oy = tile & 15;
    _Float16* P = patch[wave];
    _Float16* AW = Aim[wave];
    int half = lane >> 4, r = lane & 15;

    // ---- stage 1: zero pads, then async-load interior rows ----
    for (int t = lane; t < 960; t += 32) P[t] = (_Float16)0.0f;
    asm volatile("s_wait_dscnt 0x0" ::: "memory");

    unsigned ldsP = (unsigned)(unsigned long long)(void*)P;   // low 32 = LDS offset
    int rsub = lane >> 3;                 // 4 rows per async op
    int d = lane & 7;                     // dword within row (ix = 2d, 2d+1)
    #pragma unroll
    for (int it = 0; it < 12; ++it) {
        int row = it * 4 + rsub;          // 0..47 = c*3 + ky
        int c = row / 3, ky = row % 3;
        int iy = oy - 1 + ky;
        if (iy >= 0 && iy < 16) {
            unsigned lds = ldsP + (unsigned)(row * 40 + 4 + 4 * d);
            unsigned long long g = (unsigned long long)(const void*)
                (a1 + ((((size_t)bb * 16 + c) * 16 + iy) * 16)) + (unsigned)(4 * d);
            asm volatile("global_load_async_to_lds_b32 %0, %1, off"
                         :: "v"(lds), "v"(g) : "memory");
        }
    }
    asm volatile("s_wait_asynccnt 0x0" ::: "memory");

    // ---- stage 2: build im2col tile; k = c*9+ky*3+kx is compile-time ----
    for (int t = lane; t < 256; t += 32)                   // K-pad 144..159
        AW[(t >> 4) * 160 + 144 + (t & 15)] = (_Float16)0.0f;
    {
        int ox = r;
        #pragma unroll
        for (int cp = 0; cp < 8; ++cp) {
            int c = cp * 2 + half;                         // lane-half splits c
            const _Float16* Prow = P + c * 60;
            #pragma unroll
            for (int ky = 0; ky < 3; ++ky)
                #pragma unroll
                for (int kx = 0; kx < 3; ++kx)
                    AW[ox * 160 + c * 9 + ky * 3 + kx] = Prow[ky * 20 + ox + kx + 1];
        }
    }

    // ---- stage 3: 5 WMMAs over K=160 ----
    v8f acc = {};
    const _Float16* ap = AW + r * 160 + half * 8;
    #pragma unroll
    for (int k0 = 0; k0 < 160; k0 += 32) {
        v8h a0 = *(const v8h*)(ap + k0);
        v8h a1v = *(const v8h*)(ap + k0 + 16);
        const _Float16* bp = w2t + r * 160 + k0 + half * 8;
        v8h b0 = *(const v8h*)bp;
        v8h b1 = *(const v8h*)(bp + 16);
        v16h a, bf;
        #pragma unroll
        for (int e = 0; e < 8; ++e) {
            a[e] = a0[e]; a[8 + e] = a1v[e];
            bf[e] = b0[e]; bf[8 + e] = b1[e];
        }
        acc = __builtin_amdgcn_wmma_f32_16x16x32_f16(false, a, false, bf,
                                                     (short)0, acc, false, false);
    }
    #pragma unroll
    for (int i = 0; i < 8; ++i) {
        int ox = i + half * 8;
        y2[(((size_t)bb * 16 + r) * 16 + oy) * 16 + ox] = acc[i];
    }
}

// ------------------------------- WMMA GEMM: out = relu(A @ W^T + bias) -----
// A f16 [M,K], W f16 [N,K], bias f32. Block = 8 waves = 128 M-rows; each wave
// computes a 16 x (NT*16) strip: A frag loaded once per NT WMMAs.
template <int NT>
__global__ void kGemmWmma(const _Float16* __restrict__ A, const _Float16* __restrict__ W,
                          const float* __restrict__ bias,
                          _Float16* __restrict__ outH, float* __restrict__ outF,
                          int M, int N, int K, int nStore, int doRelu) {
    int lane = threadIdx.x & 31, wave = threadIdx.x >> 5;
    int mBase = blockIdx.x * 128 + wave * 16;
    int nBase = blockIdx.y * (NT * 16);
    int half = lane >> 4, r = lane & 15;

    const _Float16* arow = A + (size_t)(mBase + r) * K + half * 8;
    v8f acc[NT] = {};
    for (int k0 = 0; k0 < K; k0 += 32) {
        v8h a0 = *(const v8h*)(arow + k0);
        v8h a1 = *(const v8h*)(arow + k0 + 16);
        __builtin_prefetch(arow + k0 + 256, 0, 3);   // global_prefetch_b8
        v16h a;
        #pragma unroll
        for (int e = 0; e < 8; ++e) { a[e] = a0[e]; a[8 + e] = a1[e]; }
        #pragma unroll
        for (int nb = 0; nb < NT; ++nb) {
            const _Float16* brow = W + (size_t)(nBase + nb * 16 + r) * K + k0 + half * 8;
            v8h b0 = *(const v8h*)brow;
            v8h b1 = *(const v8h*)(brow + 16);
            v16h b;
            #pragma unroll
            for (int e = 0; e < 8; ++e) { b[e] = b0[e]; b[8 + e] = b1[e]; }
            acc[nb] = __builtin_amdgcn_wmma_f32_16x16x32_f16(false, a, false, b,
                                                             (short)0, acc[nb], false, false);
        }
    }
    #pragma unroll
    for (int nb = 0; nb < NT; ++nb) {
        int n = nBase + nb * 16 + r;
        float bb = (outF && n >= nStore) ? 0.0f : bias[n];
        #pragma unroll
        for (int i = 0; i < 8; ++i) {
            int m = mBase + i + half * 8;
            float v = acc[nb][i] + bb;
            if (doRelu) v = fmaxf(v, 0.0f);
            if (outH) outH[(size_t)m * N + n] = (_Float16)v;
            if (outF && n < nStore) outF[(size_t)m * nStore + n] = v;
        }
    }
}

// ---------------------------------------------------------------------------

extern "C" void kernel_launch(void* const* d_in, const int* in_sizes, int n_in,
                              void* d_out, int out_size, void* d_ws, size_t ws_size,
                              hipStream_t stream) {
    const float* x      = (const float*)d_in[0];
    const float* dfe1_w = (const float*)d_in[1];
    const float* dse1_g = (const float*)d_in[2];
    const float* dse1_b = (const float*)d_in[3];
    const float* dse1_w = (const float*)d_in[4];
    const float* bn1_g  = (const float*)d_in[5];
    const float* bn1_b  = (const float*)d_in[6];
    const float* dfe2_w = (const float*)d_in[7];
    const float* dse2_g = (const float*)d_in[8];
    const float* dse2_b = (const float*)d_in[9];
    const float* dse2_w = (const float*)d_in[10];
    const float* bn2_g  = (const float*)d_in[11];
    const float* bn2_b  = (const float*)d_in[12];
    const float* fc1_w  = (const float*)d_in[13];
    const float* fc1_b  = (const float*)d_in[14];
    const float* fc2_w  = (const float*)d_in[15];
    const float* fc2_b  = (const float*)d_in[16];
    const float* fc3_w  = (const float*)d_in[17];
    const float* fc3_b  = (const float*)d_in[18];

    const int B = 4096;
    char* ws = (char*)d_ws;

    float*    st    = (float*)ws;                               // 256 f32 stats
    _Float16* w2t   = (_Float16*)(ws + 4096);                   // 16*160
    _Float16* fc3wH = (_Float16*)(ws + 16384);                  // 32*512
    _Float16* fc1wH = (_Float16*)(ws + 65536);                  // 512*2048
    _Float16* fc2wH = (_Float16*)(ws + 65536 + 2097152);        // 512*512
    float*    y1    = (float*)(ws + (4ull << 20));              // 134MB
    _Float16* a1h   = (_Float16*)(ws + (140ull << 20));         // 33.5MB
    _Float16* a2h   = (_Float16*)(ws + (176ull << 20));         // 16.8MB
    float*    y2    = y1;                                       // reuse
    _Float16* h1    = (_Float16*)(ws + (4ull << 20));           // reuse
    _Float16* h2    = (_Float16*)(ws + (12ull << 20));

    float *sum1 = st, *sq1 = st + 8, *mean1 = st + 16, *var1 = st + 24;
    float *rsum1 = st + 32, *rsq1 = st + 40, *rmean1 = st + 48, *rvar1 = st + 56;
    float *sum2 = st + 64, *sq2 = st + 80, *mean2 = st + 96, *var2 = st + 112;
    float *rsum2 = st + 128, *rsq2 = st + 144, *rmean2 = st + 160, *rvar2 = st + 176;

    kZero<<<1, 256, 0, stream>>>(st, 256);
    kCvtF16<<<4096, 256, 0, stream>>>(fc1_w, fc1wH, 512 * 2048);
    kCvtF16<<<1024, 256, 0, stream>>>(fc2_w, fc2wH, 512 * 512);
    kCvtFc3<<<64, 256, 0, stream>>>(fc3_w, fc3wH);
    kCvtW2<<<10, 256, 0, stream>>>(dfe2_w, w2t);

    // ---- layer 1 ----
    kConv1<<<(B * 1024) / 256, 256, 0, stream>>>(x, dfe1_w, y1);
    kStats<<<dim3(1024, 8), 256, 0, stream>>>(y1, 8, 10, (size_t)B * 256, sum1, sq1);
    kFinalize<<<1, 32, 0, stream>>>(sum1, sq1, mean1, var1, 8, 1.0f / (B * 1024.0f));
    kNorm<<<dim3(1024, 8), 256, 0, stream>>>(y1, 8, 10, (size_t)B * 256, mean1, var1,
                                             dse1_g, dse1_b);
    kStats<<<dim3(1024, 8), 256, 0, stream>>>(y1, 8, 10, (size_t)B * 256, rsum1, rsq1);
    kFinalize<<<1, 32, 0, stream>>>(rsum1, rsq1, rmean1, rvar1, 8, 1.0f / (B * 1024.0f));
    kExpandPool<<<16384, 256, 0, stream>>>(y1, dse1_w, rmean1, rvar1, bn1_g, bn1_b,
                                           a1h, 8, 4, 4, (size_t)B * 16 * 256);

    // ---- layer 2 ----
    kConv2Wmma<<<(B * 16) / 4, 128, 0, stream>>>(a1h, w2t, y2);
    kStats<<<dim3(1024, 16), 256, 0, stream>>>(y2, 16, 8, (size_t)B * 64, sum2, sq2);
    kFinalize<<<1, 32, 0, stream>>>(sum2, sq2, mean2, var2, 16, 1.0f / (B * 256.0f));
    kNorm<<<dim3(1024, 16), 256, 0, stream>>>(y2, 16, 8, (size_t)B * 64, mean2, var2,
                                              dse2_g, dse2_b);
    kStats<<<dim3(1024, 16), 256, 0, stream>>>(y2, 16, 8, (size_t)B * 64, rsum2, rsq2);
    kFinalize<<<1, 32, 0, stream>>>(rsum2, rsq2, rmean2, rvar2, 16, 1.0f / (B * 256.0f));
    kExpandPool<<<8192, 256, 0, stream>>>(y2, dse2_w, rmean2, rvar2, bn2_g, bn2_b,
                                          a2h, 16, 3, 5, (size_t)B * 32 * 64);

    // ---- FC head (WMMA GEMMs) ----
    kGemmWmma<4><<<dim3(32, 8), 256, 0, stream>>>(a2h, fc1wH, fc1_b, h1, nullptr,
                                                  B, 512, 2048, 0, 1);
    kGemmWmma<4><<<dim3(32, 8), 256, 0, stream>>>(h1, fc2wH, fc2_b, h2, nullptr,
                                                  B, 512, 512, 0, 1);
    kGemmWmma<2><<<dim3(32, 1), 256, 0, stream>>>(h2, fc3wH, fc3_b, nullptr,
                                                  (float*)d_out, B, 32, 512, 10, 0);
}
